// GritMessagePassing_24824910970955
// MI455X (gfx1250) — compile-verified, matrix operand order checked by reading, not run on hardware
//
#include <hip/hip_runtime.h>

typedef __attribute__((ext_vector_type(16))) _Float16 v16h;
typedef __attribute__((ext_vector_type(8)))  _Float16 v8h;
typedef __attribute__((ext_vector_type(8)))  float    v8f;

#define N_NODES 40000
#define N_EDGES 640000
#define HID     128
#define NH      8
#define CLAMP_V 5.0f

// ---------- helpers ----------

// order-preserving float -> uint encoding (for atomicMax-based segment max)
__device__ __forceinline__ unsigned encOrd(float x) {
  unsigned u = __float_as_uint(x);
  return (u & 0x80000000u) ? ~u : (u | 0x80000000u);
}
__device__ __forceinline__ float decOrd(unsigned u) {
  return (u & 0x80000000u) ? __uint_as_float(u & 0x7FFFFFFFu) : __uint_as_float(~u);
}

// Load a 16x32 f16 A-fragment from an f32 row-major matrix.
// Wave32 A layout: lane L: M = L&15, ko = (L>>4)*8; element j <-> K = ko + (j<8 ? j : j+8).
__device__ __forceinline__ v16h load_a_frag_f32(const float* rowbase, int kb, int ko) {
  const float4* p0 = (const float4*)(rowbase + kb * 32 + ko);
  const float4* p1 = (const float4*)(rowbase + kb * 32 + ko + 16);
  float4 f0 = p0[0], f1 = p0[1], g0 = p1[0], g1 = p1[1];
  v16h a;
  a[0] = (_Float16)f0.x;  a[1] = (_Float16)f0.y;  a[2] = (_Float16)f0.z;  a[3] = (_Float16)f0.w;
  a[4] = (_Float16)f1.x;  a[5] = (_Float16)f1.y;  a[6] = (_Float16)f1.z;  a[7] = (_Float16)f1.w;
  a[8] = (_Float16)g0.x;  a[9] = (_Float16)g0.y;  a[10] = (_Float16)g0.z; a[11] = (_Float16)g0.w;
  a[12] = (_Float16)g1.x; a[13] = (_Float16)g1.y; a[14] = (_Float16)g1.z; a[15] = (_Float16)g1.w;
  return a;
}

// ---------- kernel 0: pack 6 weight matrices into f16 WMMA-B fragment order ----------
// pw index = (((mat*8 + nb)*4 + kb)*32 + lane)*16 + j ; B layout mirrors A layout with N=lane&15.
__global__ void pack_weights(const float* Qw, const float* Kw, const float* Vw,
                             const float* Eww, const float* Ebw, const float* Eow,
                             _Float16* pw) {
  int t = blockIdx.x * blockDim.x + threadIdx.x;
  const int TOT = 6 * 8 * 4 * 512;
  if (t >= TOT) return;
  int mat = t >> 14;
  int rem = t & 16383;
  int nb = rem >> 11;
  int rem2 = rem & 2047;
  int kb = rem2 >> 9;
  int idx = rem2 & 511;
  int lane = idx >> 4, j = idx & 15;
  int n = lane & 15, ko = (lane >> 4) * 8;
  int k = kb * 32 + ko + (j < 8 ? j : j + 8);
  const float* W = (mat == 0) ? Qw : (mat == 1) ? Kw : (mat == 2) ? Vw
                 : (mat == 3) ? Eww : (mat == 4) ? Ebw : Eow;
  pw[t] = (_Float16)W[k * HID + nb * 16 + n];
}

// ---------- kernel 1: zero scratch accumulators ----------
__global__ void init_ws(float* agg, float* rowV, unsigned* mxEnc, float* denom) {
  int t = blockIdx.x * blockDim.x + threadIdx.x;
  if (t < N_NODES * HID) { agg[t] = 0.0f; rowV[t] = 0.0f; }
  if (t < N_NODES * NH)  { mxEnc[t] = 0u; denom[t] = 0.0f; }
}

// ---------- kernel 2: node GEMMs Qh/Kh/Vh = x @ {Qw,Kw,Vw} ----------
__global__ __launch_bounds__(256) void node_gemm(const float* x, const _Float16* pw,
                                                 float* Qh, float* Kh, float* Vh) {
  int wave = (blockIdx.x * blockDim.x + threadIdx.x) >> 5;
  int lane = threadIdx.x & 31;
  if (wave >= N_NODES / 16) return;
  int lo = lane & 15, hi = lane >> 4, ko = hi * 8;
  int r0 = wave * 16;
  const float* rowb = x + (size_t)(r0 + lo) * HID;
  v16h a[4];
#pragma unroll
  for (int kb = 0; kb < 4; kb++) a[kb] = load_a_frag_f32(rowb, kb, ko);

  float* outs[3] = {Qh, Kh, Vh};
#pragma unroll
  for (int mat = 0; mat < 3; mat++) {
    float* O = outs[mat];
    for (int nb = 0; nb < 8; nb++) {
      v8f c = {};
#pragma unroll
      for (int kb = 0; kb < 4; kb++) {
        const v16h* bp = (const v16h*)(pw + ((size_t)((mat * 8 + nb) * 4 + kb)) * 512 + lane * 16);
        c = __builtin_amdgcn_wmma_f32_16x16x32_f16(false, a[kb], false, bp[0],
                                                   (short)0, c, false, false);
      }
#pragma unroll
      for (int r = 0; r < 8; r++)
        O[(size_t)(r0 + r + 8 * hi) * HID + nb * 16 + lo] = c[r];
    }
  }
}

// ---------- kernel 3: fused edge pipeline ----------
// Ew/Eb GEMMs -> gather Qh[dst]+Kh[src] -> signed_sqrt/relu -> (LDS refragment) ->
// Oe GEMM -> store Oe, per-head score dot w/ Aw, clamp, atomic segment-max.
__global__ __launch_bounds__(256) void edge_kernel(
    const float* conn, const long long* dst, const long long* src,
    const float* Qh, const float* Kh, const _Float16* pw,
    const float* Ebb, const float* Eob, const float* Aw,
    float* Oe, float* score, unsigned* mxEnc) {
  __shared__ _Float16 smem[8 * 16 * HID];  // 8 waves x (16 edges x 128) f16 = 32 KB
  int waveInB = threadIdx.x >> 5;
  int wave = (blockIdx.x * blockDim.x + threadIdx.x) >> 5;  // tile id (exactly 40000)
  int lane = threadIdx.x & 31;
  int lo = lane & 15, hi = lane >> 4, ko = hi * 8;
  _Float16* sm = smem + waveInB * 16 * HID;
  int e0 = wave * 16;

  // per-row edge metadata (C-fragment row M = r + 8*hi)
  int qrow[8], krow[8], didx[8];
#pragma unroll
  for (int r = 0; r < 8; r++) {
    int e = e0 + r + 8 * hi;
    int d = (int)dst[e], s = (int)src[e];
    qrow[r] = d * HID; krow[r] = s * HID; didx[r] = d * NH;
  }

  // A fragments of conn tile
  const float* rowb = conn + (size_t)(e0 + lo) * HID;
  v16h a[4];
#pragma unroll
  for (int kb = 0; kb < 4; kb++) a[kb] = load_a_frag_f32(rowb, kb, ko);

  // phase 1: Ew, Eb, elementwise, stash activation as f16 in LDS
  for (int nb = 0; nb < 8; nb++) {
    int col = nb * 16 + lo;
    v8f cw = {}, cb = {};
#pragma unroll
    for (int kb = 0; kb < 4; kb++) {
      const v16h* bw = (const v16h*)(pw + ((size_t)((3 * 8 + nb) * 4 + kb)) * 512 + lane * 16);
      const v16h* bb = (const v16h*)(pw + ((size_t)((4 * 8 + nb) * 4 + kb)) * 512 + lane * 16);
      cw = __builtin_amdgcn_wmma_f32_16x16x32_f16(false, a[kb], false, bw[0], (short)0, cw, false, false);
      cb = __builtin_amdgcn_wmma_f32_16x16x32_f16(false, a[kb], false, bb[0], (short)0, cb, false, false);
    }
    float ebb = Ebb[col];
#pragma unroll
    for (int r = 0; r < 8; r++) {
      float m1 = Qh[qrow[r] + col] + Kh[krow[r] + col];  // L2-resident gathers
      float t = m1 * cw[r];
      float ss = copysignf(sqrtf(fabsf(t)), t);          // signed_sqrt
      float act = fmaxf(ss + cb[r] + ebb, 0.0f);         // + Eb, relu
      sm[(r + 8 * hi) * HID + col] = (_Float16)act;
    }
  }
  __syncthreads();

  // reload activation tile as A fragments from LDS
#pragma unroll
  for (int kb = 0; kb < 4; kb++) {
    v8h l0 = *(const v8h*)(sm + lo * HID + kb * 32 + ko);
    v8h l1 = *(const v8h*)(sm + lo * HID + kb * 32 + ko + 16);
    a[kb] = __builtin_shufflevector(l0, l1, 0, 1, 2, 3, 4, 5, 6, 7,
                                    8, 9, 10, 11, 12, 13, 14, 15);
  }

  // phase 2: Oe GEMM, store, score (head h == column block nb, d == lo since D=16)
  for (int nb = 0; nb < 8; nb++) {
    int col = nb * 16 + lo;
    v8f c = {};
#pragma unroll
    for (int kb = 0; kb < 4; kb++) {
      const v16h* bo = (const v16h*)(pw + ((size_t)((5 * 8 + nb) * 4 + kb)) * 512 + lane * 16);
      c = __builtin_amdgcn_wmma_f32_16x16x32_f16(false, a[kb], false, bo[0], (short)0, c, false, false);
    }
    float eob = Eob[col];
    float awv = Aw[lo * NH + nb];  // Aw[d][h]
#pragma unroll
    for (int r = 0; r < 8; r++) {
      float o = c[r] + eob;
      Oe[(size_t)(e0 + r + 8 * hi) * HID + col] = o;
      float v = o * awv;
      v += __shfl_xor(v, 1);
      v += __shfl_xor(v, 2);
      v += __shfl_xor(v, 4);
      v += __shfl_xor(v, 8);  // 16-lane half-wave reduction over d
      v = fminf(fmaxf(v, -CLAMP_V), CLAMP_V);
      if (lo == 0) {
        int e = e0 + r + 8 * hi;
        score[(size_t)e * NH + nb] = v;
        atomicMax(mxEnc + didx[r] + nb, encOrd(v));
      }
    }
  }
}

// ---------- kernel 4: exp(score - mx[dst]) and denom segment-sum ----------
__global__ void softmax_norm(const long long* dst, const unsigned* mxEnc,
                             float* score, float* denom) {
  int t = blockIdx.x * blockDim.x + threadIdx.x;
  if (t >= N_EDGES * NH) return;
  int e = t >> 3, h = t & 7;
  int d = (int)dst[e];
  float ex = __expf(score[t] - decOrd(mxEnc[d * NH + h]));
  score[t] = ex;  // reuse buffer as ex
  atomicAdd(denom + d * NH + h, ex);
}

// ---------- kernel 5: weighted scatter of V[src] and ce into agg/rowV ----------
__global__ void aggregate(const long long* dst, const long long* src,
                          const float* ex, const float* denom, const float* Vh,
                          const float* Oe, float* agg, float* rowV) {
  long long t = (long long)blockIdx.x * blockDim.x + threadIdx.x;
  if (t >= (long long)N_EDGES * HID) return;
  int e = (int)(t >> 7), c = (int)(t & 127), h = c >> 4;
  int d = (int)dst[e], s = (int)src[e];
  float w = ex[e * NH + h] / (denom[d * NH + h] + 1e-16f);
  atomicAdd(agg + (size_t)d * HID + c, Vh[(size_t)s * HID + c] * w);
  atomicAdd(rowV + (size_t)d * HID + c, Oe[t] * w);
}

// ---------- kernel 6: On = agg + rowV @ BW (per-head 16x16) ----------
__global__ void finalize(const float* agg, const float* rowV, const float* BW, float* On) {
  int t = blockIdx.x * blockDim.x + threadIdx.x;
  if (t >= N_NODES * HID) return;
  int n = t >> 7, c = t & 127, h = c >> 4, dp = c & 15;
  float acc = agg[t];
  const float* rv = rowV + (size_t)n * HID + h * 16;
#pragma unroll
  for (int d = 0; d < 16; d++) acc += rv[d] * BW[d * HID + h * 16 + dp];
  On[t] = acc;
}

// ---------- launch ----------
extern "C" void kernel_launch(void* const* d_in, const int* in_sizes, int n_in,
                              void* d_out, int out_size, void* d_ws, size_t ws_size,
                              hipStream_t stream) {
  const float* x    = (const float*)d_in[0];
  const float* conn = (const float*)d_in[1];
  const long long* ei = (const long long*)d_in[2];  // int64 per reference
  const long long* dst = ei;
  const long long* src = ei + N_EDGES;
  const float* Qw  = (const float*)d_in[3];
  const float* Kw  = (const float*)d_in[4];
  const float* Vw  = (const float*)d_in[5];
  const float* Eww = (const float*)d_in[6];
  const float* Ebw = (const float*)d_in[7];
  const float* Ebb = (const float*)d_in[8];
  const float* Eow = (const float*)d_in[9];
  const float* Eob = (const float*)d_in[10];
  const float* Aw  = (const float*)d_in[11];
  const float* BW  = (const float*)d_in[12];

  float* On = (float*)d_out;                              // [N, 128]
  float* Oe = (float*)d_out + (size_t)N_NODES * HID;      // [E, 128]

  char* ws = (char*)d_ws;
  size_t off = 0;
  auto carve = [&](size_t bytes) { void* p = ws + off; off = (off + bytes + 255) & ~(size_t)255; return p; };
  float*     Qh    = (float*)carve((size_t)N_NODES * HID * 4);
  float*     Kh    = (float*)carve((size_t)N_NODES * HID * 4);
  float*     Vh    = (float*)carve((size_t)N_NODES * HID * 4);
  _Float16*  pw    = (_Float16*)carve((size_t)6 * 8 * 4 * 512 * 2);
  float*     score = (float*)carve((size_t)N_EDGES * NH * 4);
  unsigned*  mxEnc = (unsigned*)carve((size_t)N_NODES * NH * 4);
  float*     denom = (float*)carve((size_t)N_NODES * NH * 4);
  float*     agg   = (float*)carve((size_t)N_NODES * HID * 4);
  float*     rowV  = (float*)carve((size_t)N_NODES * HID * 4);

  pack_weights<<<(6 * 16384 + 255) / 256, 256, 0, stream>>>(Qw, Kw, Vw, Eww, Ebw, Eow, pw);
  init_ws<<<(N_NODES * HID + 255) / 256, 256, 0, stream>>>(agg, rowV, mxEnc, denom);
  node_gemm<<<((N_NODES / 16) * 32 + 255) / 256, 256, 0, stream>>>(x, pw, Qh, Kh, Vh);
  edge_kernel<<<(N_EDGES / 16) * 32 / 256, 256, 0, stream>>>(conn, dst, src, Qh, Kh, pw,
                                                             Ebb, Eob, Aw, Oe, score, mxEnc);
  softmax_norm<<<(N_EDGES * NH) / 256, 256, 0, stream>>>(dst, mxEnc, score, denom);
  aggregate<<<((long long)N_EDGES * HID) / 256, 256, 0, stream>>>(dst, src, score, denom,
                                                                  Vh, Oe, agg, rowV);
  finalize<<<(N_NODES * HID) / 256, 256, 0, stream>>>(agg, rowV, BW, On);
}
